// MultiRuleQuartetRankNet_4440996184281
// MI455X (gfx1250) — compile-verified, matrix operand order checked by reading, not run on hardware
//
#include <hip/hip_runtime.h>
#include <hip/hip_bf16.h>
#include <math.h>

#define BB 2048
#define KCAND 5
#define NIMG (BB*KCAND)
#define TAU 0.35f
#define RTEMP 0.7f
#define EPSV 1e-5f

typedef __attribute__((ext_vector_type(16))) __bf16 v16bf;
typedef __attribute__((ext_vector_type(8)))  __bf16 v8bf;
typedef __attribute__((ext_vector_type(8)))  float  v8f;

__device__ __forceinline__ float frcp(float x){ return __builtin_amdgcn_rcpf(x); }
__device__ __forceinline__ float frsq(float x){ return __builtin_amdgcn_rsqf(x); }

// branch-free tanh-form GELU (native v_exp + v_rcp, no libm branches)
__device__ __forceinline__ float gelu_f(float x){
  float u  = x*(0.7978845608f + 0.0356774081f*x*x);
  float t  = __expf(-2.0f*fabsf(u));
  float th = (1.0f - t) * frcp(1.0f + t);
  th = copysignf(th, u);
  return 0.5f*x*(1.0f+th);
}
__device__ __forceinline__ float sigm(float x){ return frcp(1.0f+__expf(-x)); }

__device__ __forceinline__ unsigned short f2bf(float x){
  unsigned int u = __builtin_bit_cast(unsigned int, x);
  u += 0x7fffu + ((u>>16)&1u);
  return (unsigned short)(u>>16);
}

__device__ __forceinline__ float softmin4(const float* x){
  float m=-1e30f;
  #pragma unroll
  for(int q=0;q<4;++q) m=fmaxf(m,-x[q]*(1.0f/TAU));
  float s=0.f;
  #pragma unroll
  for(int q=0;q<4;++q) s+=__expf(-x[q]*(1.0f/TAU)-m);
  return -TAU*(m+__logf(s));
}
__device__ __forceinline__ void meanvar4(const float* x, float* mo, float* vo){
  float m=0.f;
  #pragma unroll
  for(int q=0;q<4;++q) m+=x[q];
  m*=0.25f;
  float v=0.f;
  #pragma unroll
  for(int q=0;q<4;++q){ float d=x[q]-m; v+=d*d; }
  *mo=m; *vo=v*0.25f;
}

// ------------------------------------------------------------------
// Implicit-GEMM conv via bf16 WMMA.
// Input LDS layout: [(HIN+2)][(WIN+2)][CPAD] bf16, halo+channel zero-padded.
// im2col K-order: K = tap*CPAD + c  (tap = 3x3 position, c = channel).
// CDNA5 16-bit A layout => per lane two runs of 8 consecutive K
//  -> each run is one 16B-aligned ds_load_b128. No bounds checks.
// ------------------------------------------------------------------
template<int CPAD,int CIN,int COUT,int HIN,int WIN,int HOUT,int WOUT,int STRIDE,
         int OCPAD,bool OUTF32>
__device__ __forceinline__ void conv_wmma(const unsigned short* in,
                                          const unsigned short* zrun,
                                          const float* w,
                                          const float* sc, const float* sb,
                                          unsigned short* outb, float* outf,
                                          int lane, int wave)
{
  constexpr int M   = HOUT*WOUT;
  constexpr int MT  = M/16;
  constexpr int KP  = 9*CPAD;
  constexpr int NKB = (KP+31)/32;
  const int laneM = lane & 15;
  const int hiK   = (lane>>4)<<3;   // 0 / 8
  const int hiM   = hiK;

  // B fragments (weights remapped to K = tap*CPAD + c)
  v16bf Bf[NKB];
  #pragma unroll
  for (int kb=0; kb<NKB; ++kb){
    v16bf bb;
    #pragma unroll
    for (int v=0; v<8; ++v){
      #pragma unroll
      for (int t=0; t<2; ++t){
        int kk = kb*32 + ((v<4)? 2*v : 16+2*(v-4)) + hiK + t;
        int r = kk / CPAD, c = kk % CPAD;
        float val = 0.f;
        if (r < 9 && c < CIN && laneM < COUT)
          val = w[(laneM*CIN + c)*9 + r];
        bb[2*v+t] = (__bf16)val;
      }
    }
    Bf[kb] = bb;
  }

  // loop-invariant per-lane BN affine (hoisted out of tile loop)
  const float scv = sc[laneM < COUT ? laneM : 0];
  const float sbv = sb[laneM < COUT ? laneM : 0];
  const bool  wr  = (laneM < COUT);

  for (int mt = wave; mt < MT; mt += 8){
    const int m  = mt*16 + laneM;
    const int oy = m / WOUT, ox = m % WOUT;
    const int py = oy*STRIDE, px = ox*STRIDE;   // halo coords of tap (0,0)
    v8f acc = {};
    #pragma unroll
    for (int kb=0; kb<NKB; ++kb){
      v8bf run0, run1;
      #pragma unroll
      for (int rr=0; rr<2; ++rr){
        const int K0lo = kb*32 + rr*16;
        const int rA = K0lo / CPAD;        // tap for lanes 0-15
        const int rB = (K0lo+8) / CPAD;    // tap for lanes 16-31
        v8bf rv;
        if (rA >= 9 && rB >= 9){
          rv = v8bf{};                     // fully padded run (compile-time)
        } else {
          int K0 = K0lo + hiK;
          int r  = K0 / CPAD;
          int c0 = K0 - r*CPAD;
          int dy = r/3, dx = r - (r/3)*3;
          const unsigned short* p = in + ((py+dy)*(WIN+2) + (px+dx))*CPAD + c0;
          if (rA >= 9 || rB >= 9)          // mixed-validity block: addr select, no branch
            p = (r < 9) ? p : zrun;
          rv = *(const v8bf*)p;
        }
        if (rr==0) run0 = rv; else run1 = rv;
      }
      v16bf Af = __builtin_shufflevector(run0, run1,
                   0,1,2,3,4,5,6,7,8,9,10,11,12,13,14,15);
      acc = __builtin_amdgcn_wmma_f32_16x16x32_bf16(false, Af, false, Bf[kb],
                                                    (short)0, acc, false, false);
    }
    // D layout: N = lane&15, M = mt*16 + v + 8*(lane>=16)
    // compute epilogue unconditionally (packed VALU), predicate only the store
    #pragma unroll
    for (int v=0; v<8; ++v){
      int mm = mt*16 + v + hiM;
      int y = mm / WOUT, x = mm % WOUT;
      float g = gelu_f(acc[v]*scv + sbv);
      if constexpr (OUTF32){
        if (wr) outf[laneM*M + mm] = g;
      } else {
        if (wr) outb[((y+1)*(WOUT+2) + (x+1))*OCPAD + laneM] = f2bf(g);
      }
    }
  }
}

struct CnnArgs {
  const float* x;
  const float* cw[4]; const float* cb[4]; const float* cg[4];
  const float* cm[4]; const float* cv[4];
  const float* pw; const float* pb;
  float* raw;
};

__global__ __launch_bounds__(256) void k_cnn(CnnArgs a){
  // regA: buf1 [34][34][8] (18496B)  later reused as buf3 [18][18][16] (10368B) + a4 f32 [10][256] (10240B)
  __shared__ __align__(16) unsigned int regA_u32[5152];    // 20608 B
  __shared__ __align__(16) unsigned int buf2_u32[9248];    // 36992 B : [34][34][16]
  __shared__ __align__(16) float img2[34*34];              // zero-haloed input image
  __shared__ float pool[90];
  __shared__ float scs[4][16], sbs[4][16];
  __shared__ __align__(16) unsigned short zbuf[16];

  unsigned short* buf1 = (unsigned short*)regA_u32;
  unsigned short* buf3 = (unsigned short*)regA_u32;
  float*          a4   = (float*)(regA_u32 + 2592);        // offset 10368 B
  unsigned short* buf2 = (unsigned short*)buf2_u32;

  const int tid  = threadIdx.x;
  const int lane = tid & 31, wave = tid >> 5;
  const int img_id = blockIdx.x;
  const float* src = a.x + (size_t)img_id*1024;

  __builtin_prefetch(src + (tid<<2), 0, 0);
  // phase 0: zero all padded buffers
  for (int i=tid; i<5152; i+=256) regA_u32[i] = 0u;
  for (int i=tid; i<9248; i+=256) buf2_u32[i] = 0u;
  for (int i=tid; i<34*34; i+=256) img2[i] = 0.f;
  if (tid < 16) zbuf[tid] = 0;
  __syncthreads();
  // phase 1: fill image interior (1 b128 load per thread) + BN scale/bias
  {
    const float4 vx = ((const float4*)src)[tid];   // 256 float4 = 1024 floats
    int y = tid>>3, x4 = (tid&7)*4;
    float* dst = &img2[(y+1)*34 + (x4+1)];
    dst[0]=vx.x; dst[1]=vx.y; dst[2]=vx.z; dst[3]=vx.w;
  }
  if (tid < 64){
    const int CO[4] = {8,10,10,10};
    int s = tid>>4, c = tid&15;
    if (c < CO[s]){
      float scale = a.cg[s][c] * frsq(a.cv[s][c] + EPSV);
      scs[s][c] = scale;
      sbs[s][c] = a.cb[s][c] - a.cm[s][c]*scale;
    }
  }
  __syncthreads();

  // conv1: 1->8, K=9, branch-free taps on zero-haloed image
  for (int o=tid; o<8*1024; o+=256){
    int c = o>>10, p = o&1023, oy = p>>5, ox = p&31;
    const float* w = a.cw[0] + c*9;
    float s = 0.f;
    #pragma unroll
    for (int r=0; r<9; ++r)
      s += w[r]*img2[(oy + r/3)*34 + (ox + r%3)];
    buf1[((oy+1)*34 + (ox+1))*8 + c] = f2bf(gelu_f(s*scs[0][c] + sbs[0][c]));
  }
  __syncthreads();
  // conv2: 8->10, 32x32, CPAD=8, K=72 (pad 96, 3 WMMA/tile)
  conv_wmma<8,8,10,32,32,32,32,1,16,false>(buf1, zbuf, a.cw[1], scs[1], sbs[1], buf2, nullptr, lane, wave);
  __syncthreads();
  // recycle regA for buf3 (+a4): zero for halo/channel padding
  for (int i=tid; i<5152; i+=256) regA_u32[i] = 0u;
  __syncthreads();
  // conv3: 10->10, stride 2 -> 16x16, CPAD=16, K=144 (pad 160, 5 WMMA/tile)
  conv_wmma<16,10,10,32,32,16,16,2,16,false>(buf2, zbuf, a.cw[2], scs[2], sbs[2], buf3, nullptr, lane, wave);
  __syncthreads();
  // conv4: 10->10, 16x16, f32 out [10][256]
  conv_wmma<16,10,10,16,16,16,16,1,0,true>(buf3, zbuf, a.cw[3], scs[3], sbs[3], nullptr, a4, lane, wave);
  __syncthreads();

  // 3x3 overlap pool (6x6 bins: starts 0,5,10)
  if (tid < 90){
    int c = tid/9, r = tid%9, si = (r/3)*5, sj = (r%3)*5;
    float s = 0.f;
    for (int y=0;y<6;++y)
      for (int x2=0;x2<6;++x2)
        s += a4[c*256 + (si+y)*16 + (sj+x2)];
    pool[tid] = s*(1.f/36.f);
  }
  __syncthreads();
  if (tid < 8){
    float s = a.pb[tid];
    for (int j=0;j<90;++j) s += a.pw[tid*90+j]*pool[j];
    a.raw[(size_t)img_id*8 + tid] = s;
  }
}

// ---------------- meta MLP + fuse ----------------
struct FuseArgs {
  const float* meta; const float* raw;
  const float* mw1; const float* mb1; const float* mlng; const float* mlnb;
  const float* mw2; const float* mb2;
  const float* fw;  const float* fb;  const float* flng; const float* flnb;
  float* h;
};
__global__ void k_fuse(FuseArgs a){
  int i = blockIdx.x*blockDim.x + threadIdx.x;
  if (i >= NIMG) return;
  float m0[8];
  for (int j=0;j<8;++j) m0[j] = a.meta[(size_t)i*8+j];
  float t1[4];
  for (int j=0;j<4;++j){ float s=a.mb1[j]; for(int q=0;q<8;++q) s+=a.mw1[j*8+q]*m0[q]; t1[j]=gelu_f(s); }
  float mu=0; for(int j=0;j<4;++j) mu+=t1[j]; mu*=0.25f;
  float var=0; for(int j=0;j<4;++j){ float d=t1[j]-mu; var+=d*d; } var*=0.25f;
  float inv=frsq(var+EPSV);
  float m1[4]; for(int j=0;j<4;++j) m1[j]=(t1[j]-mu)*inv*a.mlng[j]+a.mlnb[j];
  float in12[12];
  for (int j=0;j<8;++j) in12[j] = a.raw[(size_t)i*8+j];
  for (int j=0;j<4;++j){ float s=a.mb2[j]; for(int q=0;q<4;++q) s+=a.mw2[j*4+q]*m1[q]; in12[8+j]=s; }
  float f[16];
  for (int j=0;j<16;++j){ float s=a.fb[j]; for(int q=0;q<12;++q) s+=a.fw[j*12+q]*in12[q]; f[j]=gelu_f(s); }
  mu=0; for(int j=0;j<16;++j) mu+=f[j]; mu*=(1.f/16.f);
  var=0; for(int j=0;j<16;++j){ float d=f[j]-mu; var+=d*d; } var*=(1.f/16.f);
  inv=frsq(var+EPSV);
  for (int j=0;j<16;++j) a.h[(size_t)i*16+j]=(f[j]-mu)*inv*a.flng[j]+a.flnb[j];
}

// ---------------- transformer (K=5, D=16, 4 heads) ----------------
struct BlockP {
  const float *ln1g,*ln1b,*qkv_w,*qkv_b,*o_w,*o_b,*ln2g,*ln2b,*fw1,*fb1,*fw2,*fb2;
};
struct XformArgs { float* h; BlockP blk[2]; };

__global__ __launch_bounds__(128) void k_xform(XformArgs a){
  __shared__ float hb[5][16], lnb_[5][16], qb[5][16], kbuf[5][16], vb[5][16], ob[5][16], hid[5][32];
  const int tid = threadIdx.x;
  const int k = tid/16, d = tid%16;   // blockDim = 80
  float* hg = a.h + (size_t)blockIdx.x*80;
  hb[k][d] = hg[k*16+d];
  __syncthreads();
  for (int bi=0; bi<2; ++bi){
    BlockP bp = a.blk[bi];
    {
      float mu=0; for(int j=0;j<16;++j) mu+=hb[k][j]; mu*=(1.f/16.f);
      float var=0; for(int j=0;j<16;++j){ float dd=hb[k][j]-mu; var+=dd*dd; } var*=(1.f/16.f);
      float inv=frsq(var+EPSV);
      lnb_[k][d]=(hb[k][d]-mu)*inv*bp.ln1g[d]+bp.ln1b[d];
    }
    __syncthreads();
    {
      float sq=bp.qkv_b[d], sk=bp.qkv_b[16+d], sv=bp.qkv_b[32+d];
      for(int j=0;j<16;++j){
        float x=lnb_[k][j];
        sq+=bp.qkv_w[d*16+j]*x;
        sk+=bp.qkv_w[(16+d)*16+j]*x;
        sv+=bp.qkv_w[(32+d)*16+j]*x;
      }
      qb[k][d]=sq; kbuf[k][d]=sk; vb[k][d]=sv;
    }
    __syncthreads();
    {
      int hh=d/4, dh=d%4;
      float s[5]; float mx=-1e30f;
      for(int j=0;j<5;++j){
        float dot=0;
        for(int t=0;t<4;++t) dot+=qb[k][hh*4+t]*kbuf[j][hh*4+t];
        s[j]=dot*0.5f; mx=fmaxf(mx,s[j]);
      }
      float se=0; for(int j=0;j<5;++j){ s[j]=__expf(s[j]-mx); se+=s[j]; }
      float o=0; for(int j=0;j<5;++j) o+=s[j]*vb[j][hh*4+dh];
      ob[k][d]=o*frcp(se);
    }
    __syncthreads();
    {
      float s=bp.o_b[d];
      for(int j=0;j<16;++j) s+=bp.o_w[d*16+j]*ob[k][j];
      hb[k][d]+=s;
    }
    __syncthreads();
    {
      float mu=0; for(int j=0;j<16;++j) mu+=hb[k][j]; mu*=(1.f/16.f);
      float var=0; for(int j=0;j<16;++j){ float dd=hb[k][j]-mu; var+=dd*dd; } var*=(1.f/16.f);
      float inv=frsq(var+EPSV);
      lnb_[k][d]=(hb[k][d]-mu)*inv*bp.ln2g[d]+bp.ln2b[d];
    }
    __syncthreads();
    {
      #pragma unroll
      for(int u=0;u<2;++u){
        int j32=d+16*u;
        float s=bp.fb1[j32];
        for(int j=0;j<16;++j) s+=bp.fw1[j32*16+j]*lnb_[k][j];
        hid[k][j32]=gelu_f(s);
      }
    }
    __syncthreads();
    {
      float s=bp.fb2[d];
      for(int j=0;j<32;++j) s+=bp.fw2[d*32+j]*hid[k][j];
      hb[k][d]+=s;
    }
    __syncthreads();
  }
  hg[k*16+d]=hb[k][d];
}

// ---------------- routing + heads ----------------
struct HeadArgs {
  const float* h; float* out;
  const float *rt_w1,*rt_b1,*rt_w2,*rt_b2;
  const float *g_w1,*g_b1,*g_w2,*g_b2;
  const float *pr_w1,*pr_b1,*pr_w2,*pr_b2;
  const float *qs_w1,*qs_b1,*qs_w2,*qs_b2;
  const float *cs_w1,*cs_b1,*cs_w2,*cs_b2;
  const float *rf_w1,*rf_b1,*rf_w2,*rf_b2;
  const float *rel_w;
};
__global__ void k_heads(HeadArgs a){
  int i = blockIdx.x*blockDim.x + threadIdx.x;
  if (i >= NIMG) return;
  int b = i/5, k = i%5;
  const float* hrow = a.h + (size_t)b*80;
  float cand[16], oth[4][16];
  for (int j=0;j<16;++j) cand[j]=hrow[k*16+j];
  for (int q=0;q<4;++q){
    int idx = q + (q>=k ? 1 : 0);
    for (int j=0;j<16;++j) oth[q][j]=hrow[idx*16+j];
  }
  float mu[16],var[16],ri[64];
  for (int j=0;j<16;++j){
    float s=0,mx=-1e30f,mn=1e30f;
    for (int q=0;q<4;++q){ float x=oth[q][j]; s+=x; mx=fmaxf(mx,x); mn=fminf(mn,x); }
    float m=s*0.25f, v=0;
    for (int q=0;q<4;++q){ float dd=oth[q][j]-m; v+=dd*dd; }
    v*=0.25f;
    mu[j]=m; var[j]=v;
    ri[j]=m; ri[16+j]=v; ri[32+j]=mx; ri[48+j]=mn;
  }
  float g1[16];
  for (int j=0;j<16;++j){ float s=a.rt_b1[j]; for(int q=0;q<64;++q) s+=a.rt_w1[j*64+q]*ri[q]; g1[j]=gelu_f(s); }
  float rl[4]; float rmx=-1e30f;
  for (int r=0;r<4;++r){ float s=a.rt_b2[r]; for(int j=0;j<16;++j) s+=a.rt_w2[r*16+j]*g1[j]; rl[r]=s*(1.0f/RTEMP); rmx=fmaxf(rmx,rl[r]); }
  float rsum=0; for(int r=0;r<4;++r){ rl[r]=__expf(rl[r]-rmx); rsum+=rl[r]; }
  float rinv = frcp(rsum);
  const int IA[6]={0,0,0,1,1,2}, IB[6]={1,2,3,2,3,3};
  float acc=0.f;
  for (int r=0;r<4;++r){
    float t16[16];
    for (int j=0;j<16;++j){ float s=a.g_b1[r*16+j]; for(int q=0;q<64;++q) s+=a.g_w1[(r*16+j)*64+q]*ri[q]; t16[j]=gelu_f(s); }
    float gate[16];
    for (int j=0;j<16;++j){ float s=a.g_b2[r*16+j]; for(int q=0;q<16;++q) s+=a.g_w2[(r*16+j)*16+q]*t16[q]; gate[j]=sigm(s); }
    float og[4][16], proto[16];
    for (int j=0;j<16;++j){
      float s=0;
      for (int q=0;q<4;++q){ og[q][j]=oth[q][j]*gate[j]; s+=og[q][j]; }
      proto[j]=s*0.25f;
    }
    float pn=0; for(int j=0;j<16;++j) pn+=proto[j]*proto[j]; pn=fmaxf(sqrtf(pn),1e-6f);
    float pninv = frcp(pn);
    float compat[4];
    for (int q=0;q<4;++q){
      float on=0,dp=0;
      for (int j=0;j<16;++j){ on+=og[q][j]*og[q][j]; dp+=og[q][j]*proto[j]; }
      on=fmaxf(sqrtf(on),1e-6f);
      compat[q]=0.5f*(dp*frcp(on)*pninv+1.0f);
    }
    float csoft = softmin4(compat);
    float cmean,cvar; meanvar4(compat,&cmean,&cvar);
    float pd[6];
    for (int p=0;p<6;++p){
      float s=0;
      for (int j=0;j<16;++j){ float dd=og[IA[p]][j]-og[IB[p]][j]; s+=dd*dd; }
      pd[p]=sqrtf(s);
    }
    float pmean=0,pmax=-1e30f;
    for (int p=0;p<6;++p){ pmean+=pd[p]; pmax=fmaxf(pmax,pd[p]); }
    pmean*=(1.f/6.f);
    float loo[4];
    for (int q=0;q<4;++q){
      float tn=0,on=0,dp=0;
      for (int j=0;j<16;++j){
        float tp=(proto[j]*4.f-og[q][j])*(1.f/3.f);
        tn+=tp*tp; on+=og[q][j]*og[q][j]; dp+=og[q][j]*tp;
      }
      tn=fmaxf(sqrtf(tn),1e-6f); on=fmaxf(sqrtf(on),1e-6f);
      loo[q]=0.5f*(dp*frcp(on*tn)+1.f);
    }
    float tsoft = softmin4(loo);
    float tmean,tvar; meanvar4(loo,&tmean,&tvar); (void)tmean;
    float ogr[4][6];
    for (int q=0;q<4;++q)
      for (int p=0;p<6;++p){
        float s=0; for(int j=0;j<16;++j) s+=a.rel_w[p*16+j]*og[q][j];
        ogr[q][p]=s;
      }
    float relp[6][6];
    for (int p=0;p<6;++p){
      float pri[24];
      for (int j=0;j<6;++j){
        float xa=ogr[IA[p]][j], xb=ogr[IB[p]][j];
        pri[j]=xa; pri[6+j]=xb; pri[12+j]=fabsf(xa-xb); pri[18+j]=xa*xb;
      }
      float hh[12];
      for (int j=0;j<12;++j){ float s=a.pr_b1[r*12+j]; for(int q=0;q<24;++q) s+=a.pr_w1[(r*12+j)*24+q]*pri[q]; hh[j]=gelu_f(s); }
      for (int o=0;o<6;++o){ float s=a.pr_b2[r*6+o]; for(int j=0;j<12;++j) s+=a.pr_w2[(r*6+o)*12+j]*hh[j]; relp[p][o]=s; }
    }
    float rmean[6],rmaxv[6],rspread=0;
    for (int o=0;o<6;++o){
      float s=0,mx=-1e30f;
      for (int p=0;p<6;++p){ s+=relp[p][o]; mx=fmaxf(mx,relp[p][o]); }
      float m=s*(1.f/6.f), v=0;
      for (int p=0;p<6;++p){ float dd=relp[p][o]-m; v+=dd*dd; }
      rmean[o]=m; rmaxv[o]=mx; rspread+=v*(1.f/6.f);
    }
    rspread*=(1.f/6.f);
    float in12[12]; for(int o=0;o<6;++o){ in12[o]=rmean[o]; in12[6+o]=rmaxv[o]; }
    float h16[16];
    for (int j=0;j<16;++j){ float s=a.qs_b1[r*16+j]; for(int q=0;q<12;++q) s+=a.qs_w1[(r*16+j)*12+q]*in12[q]; h16[j]=gelu_f(s); }
    float rs=a.qs_b2[r]; for(int j=0;j<16;++j) rs+=a.qs_w2[r*16+j]*h16[j];
    float rel_strength=sigm(rs);
    float cg[16]; float cn=0,dp2=0;
    for (int j=0;j<16;++j){ cg[j]=cand[j]*gate[j]; cn+=cg[j]*cg[j]; dp2+=cg[j]*proto[j]; }
    cn=fmaxf(sqrtf(cn),1e-6f);
    float cand_compat=0.5f*(dp2*frcp(cn)*pninv+1.f);
    float cri[24];
    for (int p=0;p<6;++p){
      float s1=0,s2=0;
      for (int j=0;j<16;++j){ s1+=a.rel_w[p*16+j]*cg[j]; s2+=a.rel_w[p*16+j]*proto[j]; }
      cri[p]=s1; cri[6+p]=s2; cri[12+p]=fabsf(s1-s2); cri[18+p]=s1*s2;
    }
    for (int j=0;j<16;++j){ float s=a.cs_b1[r*16+j]; for(int q=0;q<24;++q) s+=a.cs_w1[(r*16+j)*24+q]*cri[q]; h16[j]=gelu_f(s); }
    float cs=a.cs_b2[r]; for(int j=0;j<16;++j) cs+=a.cs_w2[r*16+j]*h16[j];
    float cand_rel=sigm(cs);
    float qq = 0.9f*csoft + 0.25f*cmean - 0.2f*cvar - 0.45f*pmean - 0.15f*pmax
             + 0.7f*tsoft - 0.2f*tvar + 0.3f*rel_strength - 0.1f*rspread;
    float rf[69];
    for (int j=0;j<16;++j){ rf[j]=mu[j]; rf[16+j]=var[j]; rf[32+j]=gate[j]; }
    for (int o=0;o<6;++o){ rf[48+o]=rmean[o]; rf[54+o]=rmaxv[o]; }
    rf[60]=csoft-cand_compat; rf[61]=pmean; rf[62]=pmax; rf[63]=cmean;
    rf[64]=tsoft; rf[65]=tvar; rf[66]=rel_strength; rf[67]=rspread; rf[68]=cand_rel;
    for (int j=0;j<16;++j){ float s=a.rf_b1[r*16+j]; for(int q=0;q<69;++q) s+=a.rf_w1[(r*16+j)*69+q]*rf[q]; h16[j]=gelu_f(s); }
    float lr=a.rf_b2[r]; for(int j=0;j<16;++j) lr+=a.rf_w2[r*16+j]*h16[j];
    float fin = qq - 0.2f*cand_compat - 0.2f*cand_rel + lr;
    acc += (rl[r]*rinv)*fin;
  }
  a.out[i]=acc;
}

// ---------------- host launch ----------------
// Input index map (jax tree flatten: dict keys sorted alphabetically):
// 0 centered_raw, 1 meta,
// 2..13 blocks[0]: f_b1,f_b2,f_w1,f_w2,ln1b,ln1g,ln2b,ln2g,o_b,o_w,qkv_b,qkv_w
// 14..25 blocks[1] (same order)
// 26..45 conv[0..3]: b,g,m,v,w each
// 46 fuse_b, 47 fuse_lnb, 48 fuse_lng, 49 fuse_w
// 50..69 heads: cs_b1,cs_b2,cs_w1,cs_w2,g_b1,g_b2,g_w1,g_w2,pr_b1,pr_b2,pr_w1,pr_w2,
//               qs_b1,qs_b2,qs_w1,qs_w2,rf_b1,rf_b2,rf_w1,rf_w2
// 70..75 meta_b1,meta_b2,meta_lnb,meta_lng,meta_w1,meta_w2
// 76 proj_b, 77 proj_w, 78 rel_w, 79 rt_b1, 80 rt_b2, 81 rt_w1, 82 rt_w2
extern "C" void kernel_launch(void* const* d_in, const int* in_sizes, int n_in,
                              void* d_out, int out_size, void* d_ws, size_t ws_size,
                              hipStream_t stream){
  auto P = [&](int i)->const float* { return (const float*)d_in[i]; };
  float* ws_raw = (float*)d_ws;                 // NIMG*8
  float* ws_h   = ws_raw + (size_t)NIMG*8;      // NIMG*16

  CnnArgs ca;
  ca.x = P(0);
  for (int c=0;c<4;++c){
    int base = 26 + c*5;
    ca.cb[c]=P(base+0); ca.cg[c]=P(base+1); ca.cm[c]=P(base+2); ca.cv[c]=P(base+3); ca.cw[c]=P(base+4);
  }
  ca.pb = P(76); ca.pw = P(77); ca.raw = ws_raw;
  k_cnn<<<NIMG, 256, 0, stream>>>(ca);

  FuseArgs fa;
  fa.meta=P(1); fa.raw=ws_raw;
  fa.mw1=P(74); fa.mb1=P(70); fa.mlng=P(73); fa.mlnb=P(72);
  fa.mw2=P(75); fa.mb2=P(71);
  fa.fw=P(49);  fa.fb=P(46);  fa.flng=P(48); fa.flnb=P(47);
  fa.h=ws_h;
  k_fuse<<<NIMG/256, 256, 0, stream>>>(fa);

  XformArgs xa; xa.h = ws_h;
  for (int bi=0; bi<2; ++bi){
    int base = 2 + bi*12;
    BlockP bp;
    bp.fb1 =P(base+0); bp.fb2 =P(base+1); bp.fw1 =P(base+2); bp.fw2 =P(base+3);
    bp.ln1b=P(base+4); bp.ln1g=P(base+5); bp.ln2b=P(base+6); bp.ln2g=P(base+7);
    bp.o_b =P(base+8); bp.o_w =P(base+9); bp.qkv_b=P(base+10); bp.qkv_w=P(base+11);
    xa.blk[bi]=bp;
  }
  k_xform<<<BB, 80, 0, stream>>>(xa);

  HeadArgs ha;
  ha.h = ws_h; ha.out = (float*)d_out;
  ha.cs_b1=P(50); ha.cs_b2=P(51); ha.cs_w1=P(52); ha.cs_w2=P(53);
  ha.g_b1 =P(54); ha.g_b2 =P(55); ha.g_w1 =P(56); ha.g_w2 =P(57);
  ha.pr_b1=P(58); ha.pr_b2=P(59); ha.pr_w1=P(60); ha.pr_w2=P(61);
  ha.qs_b1=P(62); ha.qs_b2=P(63); ha.qs_w1=P(64); ha.qs_w2=P(65);
  ha.rf_b1=P(66); ha.rf_b2=P(67); ha.rf_w1=P(68); ha.rf_w2=P(69);
  ha.rel_w=P(78);
  ha.rt_b1=P(79); ha.rt_b2=P(80); ha.rt_w1=P(81); ha.rt_w2=P(82);
  k_heads<<<NIMG/64, 64, 0, stream>>>(ha);
  (void)in_sizes; (void)n_in; (void)out_size; (void)ws_size;
}
